// HyenaFilter_14800457302411
// MI455X (gfx1250) — compile-verified, matrix operand order checked by reading, not run on hardware
//
#include <hip/hip_runtime.h>
#include <hip/hip_bf16.h>

// ---------------------------------------------------------------------------
// HyenaFilter on MI455X (gfx1250):
//   phase 1: implicit MLP filter -> k[d][j], stored REVERSED + zero-padded bf16
//   phase 2: causal depthwise conv as block-Toeplitz bf16 WMMA
//            (v_wmma_f32_16x16x32_bf16, K-unrolled x2), fused with + x*bias
// Data-movement design: all operand traffic is L2-resident (k: 12.7 MB,
// x: 200 MB with hot tiles resident in the 192MB global L2); per-WMMA LDS
// traffic reduced to 8x ds_load_b32 (A, parity-dual buffer) + 2x ds_load_b128
// (B). Filter windows staged via global_load_async_to_lds (ASYNCcnt).
// ---------------------------------------------------------------------------

#define D_MODEL 768
#define SEQ_LEN 8192
#define NBATCH  8
#define MLP_O   16
#define KS      (SEQ_LEN + 64)   // reversed filter row: [ k reversed | 64 zeros ]

typedef __attribute__((ext_vector_type(16))) __bf16 v16bf;
typedef __attribute__((ext_vector_type(8)))  float  v8f;

#if defined(__has_builtin)
#if __has_builtin(__builtin_amdgcn_global_load_async_to_lds_b32)
#define HAVE_ASYNC_LDS 1
#endif
#endif
#ifndef HAVE_ASYNC_LDS
#define HAVE_ASYNC_LDS 0
#endif

// ---------------------------------------------------------------------------
// Phase 1: k[d][j] = (sin-MLP(z[j]) @ W3[:,d] + b3[d]) * exp(-t[j]*|deltas[d]|)
// stored as kr[d*KS + (L-1-j)] (reversed so conv gathers ascend), 64-elt 0 pad.
// ---------------------------------------------------------------------------
__global__ void __launch_bounds__(64)
hyena_filter_gen(const float* __restrict__ z,      // (L,3)
                 const float* __restrict__ t,      // (L)
                 const float* __restrict__ deltas, // (D)
                 const float* __restrict__ freq,   // (16)
                 const float* __restrict__ W0, const float* __restrict__ b0,
                 const float* __restrict__ W1, const float* __restrict__ b1,
                 const float* __restrict__ W2, const float* __restrict__ b2,
                 const float* __restrict__ W3, const float* __restrict__ b3,
                 __bf16* __restrict__ kr)
{
    const int j = blockIdx.x * blockDim.x + threadIdx.x;   // time step
    const int d = blockIdx.y;                              // channel
    if (j >= SEQ_LEN) return;

    const float z0 = z[j * 3 + 0], z1 = z[j * 3 + 1], z2 = z[j * 3 + 2];

    float h0[MLP_O], h1[MLP_O], h2[MLP_O];
    #pragma unroll
    for (int o = 0; o < MLP_O; ++o)
        h0[o] = __builtin_sinf(freq[o] *
                 (z0 * W0[o] + z1 * W0[MLP_O + o] + z2 * W0[2 * MLP_O + o] + b0[o]));
    #pragma unroll
    for (int o = 0; o < MLP_O; ++o) {
        float a = b1[o];
        #pragma unroll
        for (int r = 0; r < MLP_O; ++r) a += h0[r] * W1[r * MLP_O + o];
        h1[o] = __builtin_sinf(freq[o] * a);
    }
    #pragma unroll
    for (int o = 0; o < MLP_O; ++o) {
        float a = b2[o];
        #pragma unroll
        for (int r = 0; r < MLP_O; ++r) a += h1[r] * W2[r * MLP_O + o];
        h2[o] = __builtin_sinf(freq[o] * a);
    }

    float acc = b3[d];
    #pragma unroll
    for (int r = 0; r < MLP_O; ++r) acc += h2[r] * W3[r * D_MODEL + d];

    const float decay = __builtin_expf(-t[j] * __builtin_fabsf(deltas[d]));
    kr[(size_t)d * KS + (SEQ_LEN - 1 - j)] = (__bf16)(acc * decay);
    if (j < 64)                                            // causal zero pad
        kr[(size_t)d * KS + SEQ_LEN + j] = (__bf16)0.0f;
}

// ---------------------------------------------------------------------------
// Phase 2: one wave per (channel d, output-tile-pair p); i0=32p+16, i1=32p.
//   A[m][K] = k-Toeplitz (shared by all 16 WMMA columns)
//   B cols 0..7  : x[b, t0    + K] -> tile i0, batch b
//   B cols 8..15 : x[b, t0-16 + K] -> tile i1, batch b
// Loop t0 = 0,64,128,... (two 32-lag WMMAs per staged 80-wide window) + tail.
// ---------------------------------------------------------------------------
__global__ void __launch_bounds__(32)
hyena_conv_wmma(const float*  __restrict__ x,     // (8,768,8192) f32
                const __bf16* __restrict__ kr,    // (768, KS) reversed filters
                const float*  __restrict__ bias,  // (768)
                float*        __restrict__ out)   // (8,768,8192) f32
{
    __shared__ __align__(16) __bf16 lds_k [88];          // filter window (80 used)
    __shared__ __align__(16) __bf16 lds_k2[88];          // window shifted by +1 elt
    __shared__ __align__(16) __bf16 lds_x [NBATCH][80];  // x, t in [t0-16, t0+64)

    const int lane  = threadIdx.x;        // 0..31 (wave32)
    const int p     = blockIdx.x;         // 0..255  tile pair
    const int d     = blockIdx.y;         // 0..767  channel
    const int i0    = 32 * p + 16;
    const int i1    = 32 * p;
    const int mn    = lane & 15;          // A row / B column
    const int khalf = (lane >> 4) * 8;    // CDNA5 16-bit A/B layout K-half

    const __bf16* krd = kr + (size_t)d * KS;
    const float   bd  = bias[d];

    // x staging role: 4 lanes per batch row, aligned float4 chunks
    const int xb = lane >> 2;             // batch row 0..7
    const int xs = lane & 3;              // chunk phase
    const float* xrb = x + ((size_t)xb * D_MODEL + d) * SEQ_LEN;

    // A gather: window[beta + e], beta = 15-mn+khalf; parity-dual buffer so
    // the base is always even -> pure ds_load_b32 of packed bf16 pairs.
    const int beta  = 15 - mn + khalf;    // 0..23
    const int abase = beta & ~1;
    const unsigned int* kwb =
        (const unsigned int*)((beta & 1) ? lds_k2 : lds_k) + (abase >> 1);
    // B gather: lds_x[bb][bgbase + e] / [.. + 16 + e], bases multiple of 8
    const int bb     = mn & 7;
    const int xoff   = (mn < 8) ? 16 : 0;
    const int bgbase = xoff + khalf;

    v8f acc = {};                         // 16x16 f32 accumulator (8 VGPRs)

    // ---- stage an 80-wide filter window + 8x80 x window at lag base t0 ----
    auto stage = [&](int t0, int rb0) {
        // filter window: 80 bf16 = 40 dwords, rb0 is 32-elt aligned, in-bounds
        // (rb0 >= 0, rb0+79 <= SEQ_LEN+47 < KS thanks to the 64-elt zero pad)
#if HAVE_ASYNC_LDS
        {
            const char* g = (const char*)(krd + rb0);
            char*       l = (char*)lds_k;
            __builtin_amdgcn_global_load_async_to_lds_b32(
                (int*)(g + 4 * lane), (int*)(l + 4 * lane), 0, 0);
            if (lane < 8)
                __builtin_amdgcn_global_load_async_to_lds_b32(
                    (int*)(g + 128 + 4 * lane), (int*)(l + 128 + 4 * lane), 0, 0);
        }
#else
        {
            const unsigned int* g = (const unsigned int*)(krd + rb0);
            unsigned int*       l = (unsigned int*)lds_k;
            l[lane] = g[lane];
            if (lane < 8) l[32 + lane] = g[32 + lane];
        }
#endif
        // x window: 8 rows x 80 floats -> bf16 (chunk-exact bounds: pads are
        // multiples of 4, so each float4 chunk is fully in or fully out)
        #pragma unroll
        for (int r = 0; r < 5; ++r) {
            const int c  = xs + 4 * r;            // chunk 0..19
            const int tt = t0 - 16 + 4 * c;
            float4 v = make_float4(0.f, 0.f, 0.f, 0.f);
            if (tt >= 0 && tt + 3 < SEQ_LEN)
                v = *(const float4*)(xrb + tt);
            union { __bf16 h[4]; uint2 u; } pk;
            pk.h[0] = (__bf16)v.x; pk.h[1] = (__bf16)v.y;
            pk.h[2] = (__bf16)v.z; pk.h[3] = (__bf16)v.w;
            *(uint2*)&lds_x[xb][4 * c] = pk.u;     // ds_store_b64, 8B aligned
        }
#if HAVE_ASYNC_LDS
        asm volatile("s_wait_asynccnt 0x0" ::: "memory");
#endif
        // build +1-shifted copy: lds_k2[i] = window[i+1]  (words 0..38)
        {
            const unsigned short* w  = (const unsigned short*)lds_k;
            unsigned int*         l2 = (unsigned int*)lds_k2;
            l2[lane] = (unsigned int)w[2 * lane + 1] |
                       ((unsigned int)w[2 * lane + 2] << 16);
            if (lane < 7)
                l2[32 + lane] = (unsigned int)w[2 * (32 + lane) + 1] |
                                ((unsigned int)w[2 * (32 + lane) + 2] << 16);
        }
    };

    auto gatherA = [&](int shift) -> v16bf {       // shift in window elements
        union { unsigned int u[8]; v16bf v; } af;
        const unsigned int* kw = kwb + (shift >> 1);
        #pragma unroll
        for (int i = 0; i < 4; ++i) { af.u[i] = kw[i]; af.u[4 + i] = kw[8 + i]; }
        return af.v;                               // 8x ds_load_b32
    };
    auto gatherB = [&](int shift) -> v16bf {
        union { uint4 q[2]; v16bf v; } bf_;
        bf_.q[0] = *(const uint4*)&lds_x[bb][bgbase + shift];        // ds_load_b128
        bf_.q[1] = *(const uint4*)&lds_x[bb][bgbase + shift + 16];   // ds_load_b128
        return bf_.v;
    };

    const int nblk = p + 1;               // lag blocks t0 = 0,32,...,32p
    int blk = 0;
    for (; blk + 2 <= nblk; blk += 2) {   // two 32-lag WMMAs per staged window
        const int t0  = 32 * blk;
        const int rb0 = SEQ_LEN - 32 - 32 * p + t0;
        stage(t0, rb0);
        __builtin_prefetch(krd + rb0 + 80, 0, 1);  // global_prefetch next window
        __syncthreads();
        v16bf a0 = gatherA(0),  bv0 = gatherB(0);
        v16bf a1 = gatherA(32), bv1 = gatherB(32);
        acc = __builtin_amdgcn_wmma_f32_16x16x32_bf16(
                  false, a0, false, bv0, (short)0, acc, false, false);
        acc = __builtin_amdgcn_wmma_f32_16x16x32_bf16(
                  false, a1, false, bv1, (short)0, acc, false, false);
        __syncthreads();
    }
    if (blk < nblk) {                     // odd tail: one 32-lag WMMA
        const int t0  = 32 * blk;
        const int rb0 = SEQ_LEN - 32 - 32 * p + t0;
        stage(t0, rb0);
        __syncthreads();
        acc = __builtin_amdgcn_wmma_f32_16x16x32_bf16(
                  false, gatherA(0), false, gatherB(0), (short)0, acc, false, false);
    }

    // writeback: lane owns (batch bb, tile, 8 consecutive time samples);
    // fuse + x*bias[d]
    const int    tile = (mn < 8) ? i0 : i1;
    const int    row0 = (lane >> 4) * 8;  // C/D layout: lanes 16..31 hold M+8
    const size_t base = ((size_t)bb * D_MODEL + d) * SEQ_LEN + tile + row0;
    #pragma unroll
    for (int v = 0; v < 8; ++v)
        out[base + v] = acc[v] + x[base + v] * bd;
}

// ---------------------------------------------------------------------------
// Launch
// ---------------------------------------------------------------------------
extern "C" void kernel_launch(void* const* d_in, const int* in_sizes, int n_in,
                              void* d_out, int out_size, void* d_ws, size_t ws_size,
                              hipStream_t stream)
{
    (void)in_sizes; (void)n_in; (void)out_size; (void)ws_size;

    const float* x      = (const float*)d_in[0];
    // d_in[1] = L (scalar int, == 8192, compile-time constant here)
    const float* z      = (const float*)d_in[2];
    const float* t      = (const float*)d_in[3];
    const float* deltas = (const float*)d_in[4];
    const float* freq   = (const float*)d_in[5];
    const float* W0     = (const float*)d_in[6];
    const float* b0     = (const float*)d_in[7];
    const float* W1     = (const float*)d_in[8];
    const float* b1     = (const float*)d_in[9];
    const float* W2     = (const float*)d_in[10];
    const float* b2     = (const float*)d_in[11];
    const float* W3     = (const float*)d_in[12];
    const float* b3     = (const float*)d_in[13];
    const float* bias   = (const float*)d_in[14];
    float* out = (float*)d_out;

    __bf16* kr = (__bf16*)d_ws;   // 768 * (8192+64) * 2 B = ~12.1 MB scratch

    dim3 fgrid(SEQ_LEN / 64, D_MODEL);
    hipLaunchKernelGGL(hyena_filter_gen, fgrid, dim3(64), 0, stream,
                       z, t, deltas, freq, W0, b0, W1, b1, W2, b2, W3, b3, kr);

    dim3 cgrid(SEQ_LEN / 32, D_MODEL);    // (256, 768) tile-pairs x channels
    hipLaunchKernelGGL(hyena_conv_wmma, cgrid, dim3(32), 0, stream,
                       x, kr, bias, out);
}
